// Model_41085657153917
// MI455X (gfx1250) — matrix operand order repaired and checked by reading, not compiled, not run on hardware
//
#include <hip/hip_runtime.h>
#include <cmath>

typedef __bf16 bf16;
typedef __attribute__((ext_vector_type(16))) __bf16 v16bf;
typedef __attribute__((ext_vector_type(8)))  __bf16 v8bf;
typedef __attribute__((ext_vector_type(4)))  __bf16 v4bf;
typedef __attribute__((ext_vector_type(8)))  float  v8f;
typedef __attribute__((ext_vector_type(4)))  int    v4i;

// Problem sizes (compile-time)
constexpr int QT = 1024;          // timesteps
constexpr int QB = 16;            // batch
constexpr int QH = 1024;          // hidden
constexpr int QN = 3 * QH;        // 3072 gate columns
constexpr int QM = QT * QB;       // 16384 GEMM rows
constexpr int QK = QH;            // 1024 reduction

// ---------------------------------------------------------------------------
// Async global->LDS copy (CDNA5). Guarded: falls back to sync copy if the
// builtin is absent on this toolchain.  Probe round 1 revealed the builtin's
// first param is  'AS(1) v4i*'  (printed by clang as '__device__ int vec4 *').
// ---------------------------------------------------------------------------
#if defined(__has_builtin)
#if __has_builtin(__builtin_amdgcn_global_load_async_to_lds_b128)
#define QRNN_ASYNC 1
#endif
#endif

typedef __attribute__((address_space(1))) v4i* as1_v4i_p;
typedef __attribute__((address_space(3))) v4i* as3_v4i_p;

__device__ __forceinline__ void cp_b128(const bf16* g, bf16* l) {
#ifdef QRNN_ASYNC
  __builtin_amdgcn_global_load_async_to_lds_b128(
      (as1_v4i_p)g, (as3_v4i_p)l, 0, 0);
#else
  *(v8bf*)l = *(const v8bf*)g;
#endif
}

__device__ __forceinline__ void wait_stage() {
#ifdef QRNN_ASYNC
#if __has_builtin(__builtin_amdgcn_s_wait_asynccnt)
  __builtin_amdgcn_s_wait_asynccnt(0);
#else
  asm volatile("s_wait_asynccnt 0" ::: "memory");
#endif
#endif
}

// ---------------------------------------------------------------------------
// W[l][K][N] fp32  ->  Wt[l][N][K] bf16   (tiled transpose + convert)
// grid (QN/32, QK/32, 3), block (32, 8)
// ---------------------------------------------------------------------------
__global__ void qrnn_wtrans(const float* __restrict__ W, bf16* __restrict__ Wt) {
  __shared__ float tile[32][33];
  const int l = blockIdx.z;
  const int n0 = blockIdx.x * 32, k0 = blockIdx.y * 32;
  const int tx = threadIdx.x, ty = threadIdx.y;
  const float* Wl = W + (size_t)l * QK * QN;
  bf16* Wtl = Wt + (size_t)l * QN * QK;
#pragma unroll
  for (int i = 0; i < 4; ++i)
    tile[ty + i * 8][tx] = Wl[(size_t)(k0 + ty + i * 8) * QN + n0 + tx];
  __syncthreads();
#pragma unroll
  for (int i = 0; i < 4; ++i)
    Wtl[(size_t)(n0 + ty + i * 8) * QK + k0 + tx] = (bf16)tile[tx][ty + i * 8];
}

// ---------------------------------------------------------------------------
// h0[row][k] = bf16(emb[x[row]][k]);  4 elements per thread
// grid 16384 blocks * 256 threads  (QM * QH / 4 threads)
// ---------------------------------------------------------------------------
__global__ __launch_bounds__(256) void qrnn_embed(const int* __restrict__ x,
                                                  const float* __restrict__ emb,
                                                  bf16* __restrict__ h) {
  const int idx = blockIdx.x * 256 + threadIdx.x;   // quad index
  const int row = idx >> 8;                         // QH/4 = 256 quads per row
  const int kq  = (idx & 255) << 2;
  const int tok = x[row];
  const float4 v = *(const float4*)(emb + (size_t)tok * QH + kq);
  v4bf o;
  o[0] = (bf16)v.x; o[1] = (bf16)v.y; o[2] = (bf16)v.z; o[3] = (bf16)v.w;
  *(v4bf*)(h + (size_t)row * QH + kq) = o;
}

// ---------------------------------------------------------------------------
// G = act(A[M][K] x Wt[N][K]^T + bias):  bf16 WMMA, fp32 accum, fused
// bias + tanh/sigmoid epilogue. Block tile 128x128, 8 waves (4Mx2N),
// wave tile 32x64, BK=64 double-buffered async LDS staging.
// grid (QN/128, QM/128), block 256
// ---------------------------------------------------------------------------
__global__ __launch_bounds__(256) void qrnn_gemm(const bf16* __restrict__ A,
                                                 const bf16* __restrict__ Bt,
                                                 const float* __restrict__ bias,
                                                 bf16* __restrict__ G) {
  constexpr int LDSK = 72;          // 64 K-elems + 8 pad (row = 144 B, 16B-aligned)
  constexpr int BK = 64;
  constexpr int NKC = QK / BK;      // 16 chunks

  const int tid = threadIdx.x;
  const int lane = tid & 31;
  const int wid = tid >> 5;
  const int waveM = wid & 3;        // 0..3  -> 32 rows each
  const int waveN = wid >> 2;       // 0..1  -> 64 cols each
  const int bM = blockIdx.y * 128;
  const int bN = blockIdx.x * 128;
  const int lg = lane >> 4;         // lane group (0/1)
  const int lr = lane & 15;

  __shared__ __align__(16) bf16 sA[2][128 * LDSK];
  __shared__ __align__(16) bf16 sB[2][128 * LDSK];

  v8f acc[2][4];
  const v8f vzero = {0.f, 0.f, 0.f, 0.f, 0.f, 0.f, 0.f, 0.f};
#pragma unroll
  for (int mt = 0; mt < 2; ++mt)
#pragma unroll
    for (int nt = 0; nt < 4; ++nt) acc[mt][nt] = vzero;

  auto stage = [&](int buf, int k0) {
#pragma unroll
    for (int p = 0; p < 4; ++p) {
      const int q = p * 256 + tid;  // 0..1023
      const int r = q >> 3;         // row 0..127
      const int s = q & 7;          // 16B segment 0..7
      cp_b128(A  + (size_t)(bM + r) * QK + k0 + s * 8, &sA[buf][r * LDSK + s * 8]);
      cp_b128(Bt + (size_t)(bN + r) * QK + k0 + s * 8, &sB[buf][r * LDSK + s * 8]);
    }
  };

  stage(0, 0);
  for (int kc = 0; kc < NKC; ++kc) {
    wait_stage();        // my async writes done
    __syncthreads();     // everyone's writes done
    if (kc + 1 < NKC) stage((kc + 1) & 1, (kc + 1) * BK);

    const bf16* a = sA[kc & 1];
    const bf16* b = sB[kc & 1];
#pragma unroll
    for (int ks = 0; ks < 2; ++ks) {      // two K=32 steps per chunk
      const int acol = ks * 32 + (lg << 3);   // A: K runs at +0/+8 per lane group
      const int bcol = ks * 32 + (lg << 3);   // B: K runs at +0/+8 per lane group
      v16bf af[2], bfr[4];
#pragma unroll
      for (int mt = 0; mt < 2; ++mt) {
        const int row = waveM * 32 + mt * 16 + lr;
        union { v16bf v; v8bf h[2]; } u;
        u.h[0] = *(const v8bf*)&a[row * LDSK + acol];
        u.h[1] = *(const v8bf*)&a[row * LDSK + acol + 16];
        af[mt] = u.v;
      }
#pragma unroll
      for (int nt = 0; nt < 4; ++nt) {
        const int row = waveN * 64 + nt * 16 + lr;
        union { v16bf v; v8bf h[2]; } u;
        u.h[0] = *(const v8bf*)&b[row * LDSK + bcol];
        u.h[1] = *(const v8bf*)&b[row * LDSK + bcol + 16];
        bfr[nt] = u.v;
      }
#pragma unroll
      for (int mt = 0; mt < 2; ++mt)
#pragma unroll
        for (int nt = 0; nt < 4; ++nt)
          acc[mt][nt] = __builtin_amdgcn_wmma_f32_16x16x32_bf16(
              false, af[mt], false, bfr[nt], (short)0, acc[mt][nt], false, false);
    }
    __syncthreads();
  }

  // Fused bias + activation epilogue: n<1024 -> tanh (z), else sigmoid (f,o)
#pragma unroll
  for (int mt = 0; mt < 2; ++mt) {
#pragma unroll
    for (int nt = 0; nt < 4; ++nt) {
      const int n = bN + waveN * 64 + nt * 16 + lr;
      const float bs = bias[n];
      const bool isZ = (n < QH);
      const int mBase = bM + waveM * 32 + mt * 16 + (lg << 3);
#pragma unroll
      for (int v = 0; v < 8; ++v) {
        const float g = acc[mt][nt][v] + bs;
        const float r = isZ ? tanhf(g) : (1.0f / (1.0f + __expf(-g)));
        G[(size_t)(mBase + v) * QN + n] = (bf16)r;
      }
    }
  }
}

// ---------------------------------------------------------------------------
// Sequential forget-mult scan over T per (b, ch) channel; gates are already
// activated.  Writes bf16 h for next layer, or fp32 d_out (t < T-1) on last.
// grid 64 * 256 = 16384 threads
// ---------------------------------------------------------------------------
__global__ __launch_bounds__(256) void qrnn_scan(const bf16* __restrict__ G,
                                                 bf16* __restrict__ hNext,
                                                 float* __restrict__ outF) {
  const int j = blockIdx.x * 256 + threadIdx.x;  // 0..16383
  const int b = j >> 10;
  const int ch = j & (QH - 1);
  float c = 0.f;
  for (int t = 0; t < QT; ++t) {
    const size_t base = (size_t)(t * QB + b) * QN + ch;
    const float z = (float)G[base];
    const float f = (float)G[base + QH];
    const float o = (float)G[base + 2 * QH];
    c = f * z + (1.f - f) * c;
    const float hv = o * c;
    if (hNext) {
      hNext[(size_t)(t * QB + b) * QH + ch] = (bf16)hv;
    } else if (t < QT - 1) {
      outF[(size_t)(t * QB + b) * QH + ch] = hv;
    }
  }
}

// ---------------------------------------------------------------------------
extern "C" void kernel_launch(void* const* d_in, const int* in_sizes, int n_in,
                              void* d_out, int out_size, void* d_ws, size_t ws_size,
                              hipStream_t stream) {
  const int*   x    = (const int*)d_in[0];     // [T, B]
  const float* emb  = (const float*)d_in[1];   // [V, H]
  const float* W    = (const float*)d_in[2];   // [L, H, 3H]
  const float* bias = (const float*)d_in[3];   // [L, 3H]
  float* out = (float*)d_out;                  // [T-1, B, H]

  // Workspace layout (bf16 elements): Wt | h | G  (~153 MB total)
  bf16* Wt = (bf16*)d_ws;                               // 3 * 3072 * 1024
  bf16* h  = Wt + (size_t)3 * QN * QK;                  // 16384 * 1024
  bf16* G  = h + (size_t)QM * QH;                       // 16384 * 3072

  qrnn_wtrans<<<dim3(QN / 32, QK / 32, 3), dim3(32, 8), 0, stream>>>(W, Wt);
  qrnn_embed<<<(QM * QH / 4) / 256, 256, 0, stream>>>(x, emb, h);

  for (int l = 0; l < 3; ++l) {
    qrnn_gemm<<<dim3(QN / 128, QM / 128), 256, 0, stream>>>(
        h, Wt + (size_t)l * QN * QK, bias + (size_t)l * QN, G);
    qrnn_scan<<<64, 256, 0, stream>>>(
        G, (l < 2) ? h : nullptr, (l < 2) ? nullptr : out);
  }
}
